// BertModel_68118181315212
// MI455X (gfx1250) — matrix operand order repaired and checked by reading, not compiled, not run on hardware
//
#include <hip/hip_runtime.h>
#include <cstddef>

// ---------------------------------------------------------------------------
// BERT-base forward for MI455X (gfx1250, wave32, WMMA).
// GEMMs: fp32 -> bf16 convert-on-load (vector fptrunc) ->
// v_wmma_f32_16x16x32_bf16, 128x128 block tile, double-buffered LDS.
// ---------------------------------------------------------------------------

#define BB 8
#define SS 512
#define HH 768
#define NHH 12
#define HDD 64
#define LL 12
#define FFF 3072
#define MM (BB * SS)   // 4096 rows

typedef __attribute__((ext_vector_type(16))) __bf16 v16bf;
typedef __attribute__((ext_vector_type(16))) float  v16f;
typedef __attribute__((ext_vector_type(8)))  float  v8f;

__device__ __forceinline__ unsigned short f2bf(float f) {
  unsigned u = __float_as_uint(f);
  u += 0x7FFFu + ((u >> 16) & 1u);          // round-to-nearest-even
  return (unsigned short)(u >> 16);
}

// A-matrix (16x32, 16-bit) fragment swizzle: element (m, k) -> lds index.
// lanes 0-15: M=0..15, K=0..15; lanes 16-31: K=16..31 (contiguous).
__device__ __forceinline__ int swzA(int m, int k) {
  int h = (k >> 4) & 1;
  int lane = m | (h << 4);
  int e = k & 15;
  return lane * 16 + e;
}

__device__ __forceinline__ v16bf ldbf(const unsigned short* p) {
  return *reinterpret_cast<const v16bf*>(p);
}

// Vector fp32 -> bf16 (RNE fptrunc; backend picks packed cvt) + 32B LDS store.
__device__ __forceinline__ void cvt_store(unsigned short* dst, v16f c) {
  *reinterpret_cast<v16bf*>(dst) = __builtin_convertvector(c, v16bf);
}

__device__ __forceinline__ v8f wmma_bf16(v16bf a, v16bf b, v8f c) {
  return __builtin_amdgcn_wmma_f32_16x16x32_bf16(
      false, a, false, b, (short)0, c, false, false);
}

// ---------------------------------------------------------------------------
// Tiled WMMA GEMM: C[M x N] = epi(A[M x K] @ W[K x N] + bias [+ R])
// Block 128x128, 8 waves (4 row-groups x 2 col-groups), wave tile 32x64
// (2x4 WMMA tiles), K-step 32, double-buffered LDS.
// EPI: 0 = bias, 1 = bias + exact GELU, 2 = bias + residual R.
// ---------------------------------------------------------------------------
template <int EPI>
__global__ __launch_bounds__(256) void gemm_bf16_wmma(
    const float* __restrict__ A, const float* __restrict__ W,
    const float* __restrict__ bias, const float* __restrict__ R,
    float* __restrict__ C, int Kdim, int Ndim) {
  __shared__ __attribute__((aligned(32))) unsigned short sA[2][8 * 512];
  __shared__ __attribute__((aligned(32))) unsigned short sB[2][8 * 512];

  const int tid = threadIdx.x;
  const int lane = tid & 31;
  const int wave = tid >> 5;
  const int wm = wave & 3;             // 4 row groups of 32
  const int wn = wave >> 2;            // 2 col groups of 64
  const int row0 = blockIdx.y * 128;
  const int col0 = blockIdx.x * 128;

  // --- A staging: thread owns one fragment-lane of one 16x32 A tile. ---
  // gm = global row in block (0..127), ahf selects K-half (lane 0-15 / 16-31).
  const int gm = tid >> 1;
  const int ahf = tid & 1;
  const int aOff = ((gm >> 4) << 9) + (((gm & 15) | (ahf << 4)) << 4);
  // e=0..7 -> k = ahf*8 + e ; e=8..15 -> k = 16 + ahf*8 + (e-8)
  const float* aSrc = A + (row0 + gm) * Kdim + (ahf << 3);

  // --- B staging: thread owns one fragment-lane of one 32x16 B tile. ---
  // gn = global col in block (0..127), bhf selects K-half; e=0..7 -> k = bhf*8 + e ; e=8..15 -> k = 16 + bhf*8 + (e-8).
  const int gn = tid >> 1;
  const int bhf = tid & 1;
  const int bOff = ((gn >> 4) << 9) + (((gn & 15) | (bhf << 4)) << 4);
  const float* bSrc = W + (bhf << 3) * Ndim + col0 + gn;

  v8f vz = {};
  v8f acc[2][4];
#pragma unroll
  for (int mt = 0; mt < 2; ++mt)
#pragma unroll
    for (int nt = 0; nt < 4; ++nt) acc[mt][nt] = vz;

  const int nk = Kdim >> 5;
  v16f ca, cb;

  // Prologue: stage tile 0 into buffer 0.
  {
    const float* p = aSrc;
#pragma unroll
    for (int i = 0; i < 8; ++i) ca[i] = p[i];
#pragma unroll
    for (int i = 0; i < 8; ++i) ca[8 + i] = p[16 + i];
    const float* q = bSrc;
#pragma unroll
    for (int i = 0; i < 8; ++i) cb[i] = q[i * Ndim];
#pragma unroll
    for (int i = 0; i < 8; ++i) cb[8 + i] = q[(16 + i) * Ndim];
    cvt_store(&sA[0][aOff], ca);
    cvt_store(&sB[0][bOff], cb);
  }
  __syncthreads();

  for (int kt = 0; kt < nk; ++kt) {
    const int cur = kt & 1;
    const bool more = (kt + 1) < nk;
    // Issue global loads for the next tile before the WMMAs.
    if (more) {
      const float* p = aSrc + ((kt + 1) << 5);
#pragma unroll
      for (int i = 0; i < 8; ++i) ca[i] = p[i];
#pragma unroll
      for (int i = 0; i < 8; ++i) ca[8 + i] = p[16 + i];
      const float* q = bSrc + ((kt + 1) << 5) * Ndim;
#pragma unroll
      for (int i = 0; i < 8; ++i) cb[i] = q[i * Ndim];
#pragma unroll
      for (int i = 0; i < 8; ++i) cb[8 + i] = q[(16 + i) * Ndim];
    }
    if (kt + 2 < nk) {
      __builtin_prefetch(aSrc + ((kt + 2) << 5), 0, 0);
      __builtin_prefetch(bSrc + ((kt + 2) << 5) * Ndim, 0, 0);
    }

    v16bf aF0 = ldbf(&sA[cur][(((wm << 1) + 0) << 9) + lane * 16]);
    v16bf aF1 = ldbf(&sA[cur][(((wm << 1) + 1) << 9) + lane * 16]);
    v16bf bF0 = ldbf(&sB[cur][(((wn << 2) + 0) << 9) + lane * 16]);
    v16bf bF1 = ldbf(&sB[cur][(((wn << 2) + 1) << 9) + lane * 16]);
    v16bf bF2 = ldbf(&sB[cur][(((wn << 2) + 2) << 9) + lane * 16]);
    v16bf bF3 = ldbf(&sB[cur][(((wn << 2) + 3) << 9) + lane * 16]);

    acc[0][0] = wmma_bf16(aF0, bF0, acc[0][0]);
    acc[0][1] = wmma_bf16(aF0, bF1, acc[0][1]);
    acc[0][2] = wmma_bf16(aF0, bF2, acc[0][2]);
    acc[0][3] = wmma_bf16(aF0, bF3, acc[0][3]);
    acc[1][0] = wmma_bf16(aF1, bF0, acc[1][0]);
    acc[1][1] = wmma_bf16(aF1, bF1, acc[1][1]);
    acc[1][2] = wmma_bf16(aF1, bF2, acc[1][2]);
    acc[1][3] = wmma_bf16(aF1, bF3, acc[1][3]);

    if (more) {
      cvt_store(&sA[cur ^ 1][aOff], ca);
      cvt_store(&sB[cur ^ 1][bOff], cb);
    }
    __syncthreads();
  }

  // Epilogue. C fragment: m = r + 8*(lane>=16), n = lane&15.
  const int mh = lane >> 4;
  const int n16 = lane & 15;
  const int colW = col0 + (wn << 6) + n16;
  float bs[4];
#pragma unroll
  for (int nt = 0; nt < 4; ++nt) bs[nt] = bias[colW + (nt << 4)];

#pragma unroll
  for (int mt = 0; mt < 2; ++mt) {
    const int mrow = row0 + (wm << 5) + (mt << 4) + (mh << 3);
#pragma unroll
    for (int nt = 0; nt < 4; ++nt) {
      const int base = mrow * Ndim + colW + (nt << 4);
      float* cp = C + base;
      const float* rp = R + base;
#pragma unroll
      for (int r = 0; r < 8; ++r) {
        float v = acc[mt][nt][r] + bs[nt];
        if (EPI == 1) {
          v = 0.5f * v * (1.0f + erff(v * 0.70710678118654752f));
        } else if (EPI == 2) {
          v += rp[r * Ndim];
        }
        cp[r * Ndim] = v;
      }
    }
  }
}

// ---------------------------------------------------------------------------
// Flash-style fused attention. grid = (S/64, NH, B), 128 threads (4 waves).
// Wave w handles 16 query rows; online softmax over 32-key blocks.
// ---------------------------------------------------------------------------
__global__ __launch_bounds__(128) void flash_attn(
    const float* __restrict__ Q, const float* __restrict__ K,
    const float* __restrict__ V, const float* __restrict__ amask,
    float* __restrict__ CTX) {
  __shared__ __attribute__((aligned(32))) unsigned short sK[4 * 512];
  __shared__ __attribute__((aligned(32))) unsigned short sV[4 * 512];
  __shared__ __attribute__((aligned(32))) unsigned short sP[4 * 512];

  const int tid = threadIdx.x;
  const int lane = tid & 31;
  const int wave = tid >> 5;
  const int hlf = lane >> 4;
  const int n16 = lane & 15;
  const int qb = blockIdx.x;
  const int head = blockIdx.y;
  const int b = blockIdx.z;
  const int qr0 = qb * 64 + wave * 16;
  const int rowBase = b * SS;
  const int colBase = head * HDD;

  // Q fragments (A-matrix 16x32 each) gathered straight into registers.
  v16bf qF0, qF1;
  {
    const float* qrow = Q + (rowBase + qr0 + n16) * HH + colBase;
    v16f c0, c1;
#pragma unroll
    for (int e = 0; e < 16; ++e) {
      int r = e >> 1;
      int kl = e + (hlf << 4);
      c0[e] = qrow[kl];
      c1[e] = qrow[32 + kl];
    }
    qF0 = __builtin_convertvector(c0, v16bf);
    qF1 = __builtin_convertvector(c1, v16bf);
  }

  // Staging map: each thread owns fragment-lane (frag = wave, lane) of sK & sV.
  // sK frag f = ktile*2+dstep (scores B: n=key, k=d_local 0..31):
  const int kKey = ((wave >> 1) << 4) + n16;           // key in 32-block
  const int kDb  = ((wave & 1) << 5) + (hlf << 4);     // d base (16 contiguous)
  const float* kSrc = K + (rowBase + kKey) * HH + colBase + kDb;
  // sV frag j (ctx B: n=d_local, k=kk 0..31):
  const int vD  = (wave << 4) + n16;                   // d
  const int vKb = hlf << 4;                            // kk base
  const float* vSrc = V + (rowBase + vKb) * HH + colBase + vD;
  const int fragOff = (wave << 9) + (lane << 4);

  v8f vz = {};
  v8f o[4];
  o[0] = vz; o[1] = vz; o[2] = vz; o[3] = vz;
  float mrow[8], lrow[8];
#pragma unroll
  for (int r = 0; r < 8; ++r) { mrow[r] = -1e30f; lrow[r] = 0.0f; }

  const float SC = 0.125f;  // 1/sqrt(64)

  for (int kb = 0; kb < SS / 32; ++kb) {
    const int key0 = kb * 32;
    // Stage K (16 contiguous floats) and V (16 strided rows).
    {
      v16f ck, cv;
      const float* p = kSrc + key0 * HH;
#pragma unroll
      for (int i = 0; i < 16; ++i) ck[i] = p[i];
      const float* q = vSrc + key0 * HH;
#pragma unroll
      for (int i = 0; i < 16; ++i) cv[i] = q[i * HH];
      cvt_store(&sK[fragOff], ck);
      cvt_store(&sV[fragOff], cv);
    }
    __syncthreads();

    v8f s0 = vz, s1 = vz;
    s0 = wmma_bf16(qF0, ldbf(&sK[0 * 512 + lane * 16]), s0);
    s0 = wmma_bf16(qF1, ldbf(&sK[1 * 512 + lane * 16]), s0);
    s1 = wmma_bf16(qF0, ldbf(&sK[2 * 512 + lane * 16]), s1);
    s1 = wmma_bf16(qF1, ldbf(&sK[3 * 512 + lane * 16]), s1);

    float madd0 = (1.0f - amask[b * SS + key0 + n16]) * -10000.0f;
    float madd1 = (1.0f - amask[b * SS + key0 + 16 + n16]) * -10000.0f;

    float p0[8], p1[8], corr[8];
#pragma unroll
    for (int r = 0; r < 8; ++r) {
      float a0 = s0[r] * SC + madd0;
      float a1 = s1[r] * SC + madd1;
      float tm = fmaxf(a0, a1);
#pragma unroll
      for (int ofs = 8; ofs >= 1; ofs >>= 1)
        tm = fmaxf(tm, __shfl_xor(tm, ofs, 32));
      float mnew = fmaxf(mrow[r], tm);
      float c = __expf(mrow[r] - mnew);
      float e0 = __expf(a0 - mnew);
      float e1 = __expf(a1 - mnew);
      float rs = e0 + e1;
#pragma unroll
      for (int ofs = 8; ofs >= 1; ofs >>= 1)
        rs += __shfl_xor(rs, ofs, 32);
      lrow[r] = lrow[r] * c + rs;
      mrow[r] = mnew;
      corr[r] = c;
      p0[r] = e0;
      p1[r] = e1;
    }
#pragma unroll
    for (int j = 0; j < 4; ++j)
#pragma unroll
      for (int r = 0; r < 8; ++r) o[j][r] *= corr[r];

    // P (16x32, C layout) -> LDS in A-fragment layout (lane transpose).
#pragma unroll
    for (int r = 0; r < 8; ++r) {
      int m = r + (hlf << 3);
      sP[(wave << 9) + swzA(m, n16)] = f2bf(p0[r]);
      sP[(wave << 9) + swzA(m, 16 + n16)] = f2bf(p1[r]);
    }
#if __has_builtin(__builtin_amdgcn_s_wait_dscnt)
    __builtin_amdgcn_s_wait_dscnt(0);
#endif
    v16bf pF = ldbf(&sP[(wave << 9) + lane * 16]);
    o[0] = wmma_bf16(pF, ldbf(&sV[0 * 512 + lane * 16]), o[0]);
    o[1] = wmma_bf16(pF, ldbf(&sV[1 * 512 + lane * 16]), o[1]);
    o[2] = wmma_bf16(pF, ldbf(&sV[2 * 512 + lane * 16]), o[2]);
    o[3] = wmma_bf16(pF, ldbf(&sV[3 * 512 + lane * 16]), o[3]);
    __syncthreads();
  }

  // Normalize and write context (fp32).
#pragma unroll
  for (int r = 0; r < 8; ++r) {
    float inv = 1.0f / lrow[r];
    int row = rowBase + qr0 + r + (hlf << 3);
#pragma unroll
    for (int j = 0; j < 4; ++j) {
      CTX[row * HH + colBase + (j << 4) + n16] = o[j][r] * inv;
    }
  }
}

// ---------------------------------------------------------------------------
// Row LayerNorm over H=768, one block per row, 256 threads x 3 elements.
// ---------------------------------------------------------------------------
__global__ __launch_bounds__(256) void layernorm768(
    const float* __restrict__ in, const float* __restrict__ g,
    const float* __restrict__ bt, float* __restrict__ out) {
  __shared__ float sh1[8], sh2[8];
  const int tid = threadIdx.x;
  const int lane = tid & 31;
  const int wave = tid >> 5;
  const int row = blockIdx.x;
  const float* p = in + row * HH;
  float x0 = p[tid], x1 = p[tid + 256], x2 = p[tid + 512];
  float s = x0 + x1 + x2;
  float q = x0 * x0 + x1 * x1 + x2 * x2;
#pragma unroll
  for (int ofs = 16; ofs >= 1; ofs >>= 1) {
    s += __shfl_xor(s, ofs, 32);
    q += __shfl_xor(q, ofs, 32);
  }
  if (lane == 0) { sh1[wave] = s; sh2[wave] = q; }
  __syncthreads();
  float ts = 0.0f, tq = 0.0f;
#pragma unroll
  for (int i = 0; i < 8; ++i) { ts += sh1[i]; tq += sh2[i]; }
  float mean = ts * (1.0f / HH);
  float var = tq * (1.0f / HH) - mean * mean;
  float rstd = rsqrtf(var + 1e-12f);
  float* o = out + row * HH;
  o[tid]       = (x0 - mean) * rstd * g[tid]       + bt[tid];
  o[tid + 256] = (x1 - mean) * rstd * g[tid + 256] + bt[tid + 256];
  o[tid + 512] = (x2 - mean) * rstd * g[tid + 512] + bt[tid + 512];
}

// ---------------------------------------------------------------------------
// Embedding gather (word + pos + type) fused with LayerNorm.
// ---------------------------------------------------------------------------
__global__ __launch_bounds__(256) void embed_ln(
    const int* __restrict__ ids, const int* __restrict__ tts,
    const float* __restrict__ we, const float* __restrict__ pe,
    const float* __restrict__ te, const float* __restrict__ g,
    const float* __restrict__ bt, float* __restrict__ X) {
  __shared__ float sh1[8], sh2[8];
  const int tid = threadIdx.x;
  const int lane = tid & 31;
  const int wave = tid >> 5;
  const int row = blockIdx.x;
  const int sidx = row & (SS - 1);
  const int id = ids[row];
  const int tt = tts[row];
  const float* wp = we + (size_t)id * HH;
  const float* pp = pe + (size_t)sidx * HH;
  const float* tp = te + (size_t)tt * HH;
  float x0 = wp[tid] + pp[tid] + tp[tid];
  float x1 = wp[tid + 256] + pp[tid + 256] + tp[tid + 256];
  float x2 = wp[tid + 512] + pp[tid + 512] + tp[tid + 512];
  float s = x0 + x1 + x2;
  float q = x0 * x0 + x1 * x1 + x2 * x2;
#pragma unroll
  for (int ofs = 16; ofs >= 1; ofs >>= 1) {
    s += __shfl_xor(s, ofs, 32);
    q += __shfl_xor(q, ofs, 32);
  }
  if (lane == 0) { sh1[wave] = s; sh2[wave] = q; }
  __syncthreads();
  float ts = 0.0f, tq = 0.0f;
#pragma unroll
  for (int i = 0; i < 8; ++i) { ts += sh1[i]; tq += sh2[i]; }
  float mean = ts * (1.0f / HH);
  float var = tq * (1.0f / HH) - mean * mean;
  float rstd = rsqrtf(var + 1e-12f);
  float* o = X + (size_t)row * HH;
  o[tid]       = (x0 - mean) * rstd * g[tid]       + bt[tid];
  o[tid + 256] = (x1 - mean) * rstd * g[tid + 256] + bt[tid + 256];
  o[tid + 512] = (x2 - mean) * rstd * g[tid + 512] + bt[tid + 512];
}

// ---------------------------------------------------------------------------
// Pooler: pooled[b] = tanh(x[b, 0] @ pool_W + pool_b).  One block per batch.
// ---------------------------------------------------------------------------
__global__ __launch_bounds__(256) void pooler(
    const float* __restrict__ X, const float* __restrict__ Wp,
    const float* __restrict__ bp, float* __restrict__ out) {
  __shared__ float xrow[HH];
  const int tid = threadIdx.x;
  const int b = blockIdx.x;
  const float* xp = X + (size_t)b * SS * HH;  // row s=0 of batch b
  xrow[tid] = xp[tid];
  xrow[tid + 256] = xp[tid + 256];
  xrow[tid + 512] = xp[tid + 512];
  __syncthreads();
  for (int n = tid; n < HH; n += 256) {
    float acc = 0.0f;
    for (int k = 0; k < HH; ++k) acc += xrow[k] * Wp[(size_t)k * HH + n];
    out[(size_t)b * HH + n] = tanhf(acc + bp[n]);
  }
}

// ---------------------------------------------------------------------------
extern "C" void kernel_launch(void* const* d_in, const int* in_sizes, int n_in,
                              void* d_out, int out_size, void* d_ws,
                              size_t ws_size, hipStream_t stream) {
  (void)in_sizes; (void)n_in; (void)out_size; (void)ws_size;

  const int*   ids   = (const int*)d_in[0];
  const int*   tts   = (const int*)d_in[1];
  const float* amask = (const float*)d_in[2];
  const float* we    = (const float*)d_in[3];
  const float* pe    = (const float*)d_in[4];
  const float* te    = (const float*)d_in[5];
  const float* eg    = (const float*)d_in[6];
  const float* eb    = (const float*)d_in[7];
  const float* Wq    = (const float*)d_in[8];
  const float* bq    = (const float*)d_in[9];
  const float* Wk    = (const float*)d_in[10];
  const float* bk    = (const float*)d_in[11];
  const float* Wv    = (const float*)d_in[12];
  const float* bv    = (const float*)d_in[13];
  const float* Wo    = (const float*)d_in[14];
  const float* bo    = (const float*)d_in[15];
  const float* g1    = (const float*)d_in[16];
  const float* b1    = (const float*)d_in[17];
  const float* Wi    = (const float*)d_in[18];
  const float* bi    = (const float*)d_in[19];
  const float* Wout  = (const float*)d_in[20];
  const float* bout  = (const float*)d_in[21];
  const float* g2    = (const float*)d_in[22];
  const float* b2    = (const float*)d_in[23];
  const float* pW    = (const float*)d_in[24];
  const float* pb    = (const float*)d_in[25];

  const size_t MH = (size_t)MM * HH;
  float* ws   = (float*)d_ws;
  float* X    = ws;
  float* Qb   = X + MH;
  float* Kb   = Qb + MH;
  float* Vb   = Kb + MH;
  float* CTXb = Vb + MH;
  float* Tb   = CTXb + MH;
  float* AOb  = Tb + MH;
  float* INT  = AOb + MH;              // M x FF

  float* outX    = (float*)d_out;      // [B,S,H]
  float* outPool = outX + MH;          // [B,H]

  embed_ln<<<MM, 256, 0, stream>>>(ids, tts, we, pe, te, eg, eb, X);

  const dim3 gH(HH / 128, MM / 128);    // (6, 32)
  const dim3 gFF(FFF / 128, MM / 128);  // (24, 32)
  const dim3 gAttn(SS / 64, NHH, BB);   // (8, 12, 8)

  for (int l = 0; l < LL; ++l) {
    const size_t oHH = (size_t)l * HH * HH;
    gemm_bf16_wmma<0><<<gH, 256, 0, stream>>>(X, Wq + oHH, bq + l * HH,
                                              nullptr, Qb, HH, HH);
    gemm_bf16_wmma<0><<<gH, 256, 0, stream>>>(X, Wk + oHH, bk + l * HH,
                                              nullptr, Kb, HH, HH);
    gemm_bf16_wmma<0><<<gH, 256, 0, stream>>>(X, Wv + oHH, bv + l * HH,
                                              nullptr, Vb, HH, HH);
    flash_attn<<<gAttn, 128, 0, stream>>>(Qb, Kb, Vb, amask, CTXb);
    gemm_bf16_wmma<2><<<gH, 256, 0, stream>>>(CTXb, Wo + oHH, bo + l * HH, X,
                                              Tb, HH, HH);
    layernorm768<<<MM, 256, 0, stream>>>(Tb, g1 + l * HH, b1 + l * HH, AOb);
    gemm_bf16_wmma<1><<<gFF, 256, 0, stream>>>(AOb, Wi + (size_t)l * HH * FFF,
                                               bi + (size_t)l * FFF, nullptr,
                                               INT, HH, FFF);
    gemm_bf16_wmma<2><<<gH, 256, 0, stream>>>(INT, Wout + (size_t)l * FFF * HH,
                                              bout + l * HH, AOb, Tb, FFF, HH);
    float* dst = (l == LL - 1) ? outX : X;
    layernorm768<<<MM, 256, 0, stream>>>(Tb, g2 + l * HH, b2 + l * HH, dst);
  }

  pooler<<<BB, 256, 0, stream>>>(outX, pW, pb, outPool);
}
